// GNN_15676630631283
// MI455X (gfx1250) — compile-verified, matrix operand order checked by reading, not run on hardware
//
#include <hip/hip_runtime.h>
#include <math.h>

typedef __attribute__((ext_vector_type(16))) _Float16 v16h;
typedef __attribute__((ext_vector_type(8)))  _Float16 v8h;
typedef __attribute__((ext_vector_type(8)))  float    v8f;

#define NN    100000
#define EGG   600000
#define EE    600000
#define KIN   130
#define KPAD  160
#define NHID  520
#define NPAD  528
#define NTN   33     // NPAD/16

// ---------- helpers ----------
__device__ __forceinline__ float eluf(float v) {
    return v > 0.f ? v : (expf(v) - 1.f);
}
// Branchless fast ELU for the hot WMMA epilogue: exp arg clamped <= 0 so the
// hardware v_exp_f32 path needs no range fixup; selects with v_cndmask.
__device__ __forceinline__ float eluf_fast(float v) {
    float e = __expf(fminf(v, 0.f)) - 1.f;
    return v > 0.f ? v : e;
}
__device__ __forceinline__ float lrelu02(float v) {
    return v > 0.f ? v : 0.2f * v;
}
__device__ __forceinline__ void atomicMaxF(float* addr, float v) {
    if (v >= 0.f) atomicMax((int*)addr, __float_as_int(v));
    else          atomicMin((unsigned int*)addr, (unsigned int)__float_as_int(v));
}

// ---------- weight prep: Wm1^T -> f16 padded [NPAD x KPAD], interleave bm1/Wm2 ----------
__global__ void prep_weights(const float* __restrict__ Wm1, const float* __restrict__ bm1,
                             const float* __restrict__ Wm2,
                             _Float16* __restrict__ w1t, float2* __restrict__ bw) {
    int i = blockIdx.x * blockDim.x + threadIdx.x;
    if (i < NPAD * KPAD) {
        int n = i / KPAD, k = i - n * KPAD;
        float v = (n < NHID && k < KIN) ? Wm1[k * NHID + n] : 0.f;
        w1t[i] = (_Float16)v;
    }
    if (i < NPAD) {
        float2 p;
        p.x = (i < NHID) ? bm1[i] : 0.f;   // hidden bias
        p.y = (i < NHID) ? Wm2[i] : 0.f;   // output weight
        bw[i] = p;
    }
}

// ---------- init accumulators ----------
__global__ void init_agg(float* __restrict__ aggO, float* __restrict__ amax,
                         float* __restrict__ denom) {
    int i = blockIdx.x * blockDim.x + threadIdx.x;
    if (i < NN * 64) aggO[i] = 0.f;
    if (i < NN) { amax[i] = -INFINITY; denom[i] = 0.f; }
}

// ---------- GAT layer 0 transform (in=3) + attention scalars ----------
__global__ void gat0_xform(const float* __restrict__ pos, const float* __restrict__ W0,
                           const float* __restrict__ asv, const float* __restrict__ adv,
                           float* __restrict__ h, float* __restrict__ a_s,
                           float* __restrict__ a_d) {
    int n = blockIdx.x * blockDim.x + threadIdx.x;
    if (n >= NN) return;
    float p0 = pos[n * 3 + 0], p1 = pos[n * 3 + 1], p2 = pos[n * 3 + 2];
    float as = 0.f, ad = 0.f;
    #pragma unroll 8
    for (int c = 0; c < 64; ++c) {
        float hv = p0 * W0[c] + p1 * W0[64 + c] + p2 * W0[128 + c];
        h[n * 64 + c] = hv;
        as += hv * asv[c];
        ad += hv * adv[c];
    }
    a_s[n] = as; a_d[n] = ad;
}

// ---------- GAT layer 1 transform (in=64): 4 nodes / 256-thread block ----------
__global__ __launch_bounds__(256) void gat1_xform(const float* __restrict__ x,
                                                  const float* __restrict__ W,
                                                  float* __restrict__ h) {
    __shared__ float sx[4][64];
    int t = threadIdx.x, nl = t >> 6, c = t & 63;
    int n = blockIdx.x * 4 + nl;
    sx[nl][c] = (n < NN) ? x[n * 64 + c] : 0.f;
    __syncthreads();
    if (n >= NN) return;
    float s = 0.f;
    #pragma unroll 8
    for (int k = 0; k < 64; ++k) s += sx[nl][k] * W[k * 64 + c];
    h[n * 64 + c] = s;
}

__global__ void node_att(const float* __restrict__ h, const float* __restrict__ asv,
                         const float* __restrict__ adv, float* __restrict__ a_s,
                         float* __restrict__ a_d) {
    int n = blockIdx.x * blockDim.x + threadIdx.x;
    if (n >= NN) return;
    float as = 0.f, ad = 0.f;
    #pragma unroll 8
    for (int c = 0; c < 64; ++c) {
        float v = h[n * 64 + c];
        as += v * asv[c];
        ad += v * adv[c];
    }
    a_s[n] = as; a_d[n] = ad;
}

// ---------- scatter-softmax passes ----------
__global__ void edge_alpha_max(const float* __restrict__ a_s, const float* __restrict__ a_d,
                               const int* __restrict__ src, const int* __restrict__ dstI,
                               float* __restrict__ amax) {
    int i = blockIdx.x * blockDim.x + threadIdx.x;
    if (i >= EGG) return;
    int d = dstI[i];
    atomicMaxF(&amax[d], lrelu02(a_s[src[i]] + a_d[d]));
}

__global__ void amax_fix(float* __restrict__ amax) {
    int n = blockIdx.x * blockDim.x + threadIdx.x;
    if (n >= NN) return;
    float a = amax[n];
    if (!(a >= -3.4e38f && a <= 3.4e38f)) amax[n] = 0.f;  // -inf (no in-edges) or NaN -> 0
}

__global__ void edge_expsum(const float* __restrict__ a_s, const float* __restrict__ a_d,
                            const int* __restrict__ src, const int* __restrict__ dstI,
                            const float* __restrict__ amax, float* __restrict__ ew,
                            float* __restrict__ denom) {
    int i = blockIdx.x * blockDim.x + threadIdx.x;
    if (i >= EGG) return;
    int d = dstI[i];
    float e = expf(lrelu02(a_s[src[i]] + a_d[d]) - amax[d]);
    ew[i] = e;
    atomicAdd(&denom[d], e);
}

__global__ void edge_agg(const int* __restrict__ src, const int* __restrict__ dstI,
                         const float* __restrict__ ew, const float* __restrict__ denom,
                         const float* __restrict__ h, float* __restrict__ aggO) {
    long long i = (long long)blockIdx.x * blockDim.x + threadIdx.x;
    if (i >= (long long)EGG * 64) return;
    int e = (int)(i >> 6), c = (int)(i & 63);
    int d = dstI[e];
    float w = ew[e] / (denom[d] + 1e-16f);
    atomicAdd(&aggO[d * 64 + c], h[src[e] * 64 + c] * w);
}

__global__ void finish_elu(const float* __restrict__ aggO, const float* __restrict__ b,
                           float* __restrict__ x) {
    int i = blockIdx.x * blockDim.x + threadIdx.x;
    if (i >= NN * 64) return;
    x[i] = eluf(aggO[i] + b[i & 63]);
}

// ---------- fused edge MLP: gather-concat -> f16 WMMA GEMM -> ELU -> dot(Wm2) ----------
// Block = 256 threads = 8 waves; block stages Wm1^T (f16, [NPAD x KPAD]) in LDS.
// Each wave owns a 32-edge tile (two M=16 sub-tiles) so every B fragment feeds
// TWO independent back-to-back WMMAs. 33 n-tiles x 5 k-steps x 2.
__global__ __launch_bounds__(256) void edge_mlp_wmma(
    const float* __restrict__ x, const int* __restrict__ esrc, const int* __restrict__ edst,
    const float* __restrict__ eattr, const _Float16* __restrict__ w1t,
    const float2* __restrict__ bw, const float* __restrict__ bm2,
    float* __restrict__ out, int nE, int nTiles) {
    extern __shared__ char smem[];
    _Float16* sB = (_Float16*)smem;                          // NPAD*KPAD halves
    _Float16* sA = (_Float16*)(smem + NPAD * KPAD * 2);      // 8 waves * 32*KPAD halves

    const int tid = threadIdx.x;
    // cooperative copy of B panel (hot in L2: 165 KB, reused by every block)
    {
        const uint4* s = (const uint4*)w1t;
        uint4* d = (uint4*)sB;
        for (int i = tid; i < NPAD * KPAD * 2 / 16; i += 256) d[i] = s[i];
    }

    const int w = tid >> 5;
    const int lane = tid & 31;
    const int tile = blockIdx.x * 8 + w;      // 32 edges per tile
    const bool valid = (tile < nTiles);

    // stage A tile: 32 edges x KPAD f16 (gather x[src] | x[dst] | attr | zero-pad)
    _Float16* a = sA + w * (32 * KPAD);
    for (int idx = lane; idx < 32 * KPAD; idx += 32) {
        int m = idx / KPAD, k = idx - m * KPAD;
        int e = tile * 32 + m;
        float v = 0.f;
        if (valid && e < nE) {
            if (k < 64)       v = x[esrc[e] * 64 + k];
            else if (k < 128) v = x[edst[e] * 64 + (k - 64)];
            else if (k < KIN) v = eattr[e * 2 + (k - 128)];
        }
        a[idx] = (_Float16)v;
    }
    __syncthreads();
    if (!valid) return;  // wave-uniform

    const int mn = lane & 15;   // A: row m; B: col n; C: n
    const int kh = lane >> 4;   // K-half select per ISA fragment layout

    // preload both A sub-tiles' fragments into registers (held across nt loop)
    v16h af0[5], af1[5];
    #pragma unroll
    for (int ks = 0; ks < 5; ++ks) {
        const v8h* p0 = (const v8h*)(a + mn * KPAD + ks * 32 + kh * 8);
        const v8h* p1 = (const v8h*)(a + (16 + mn) * KPAD + ks * 32 + kh * 8);
        ((v8h*)&af0[ks])[0] = p0[0]; ((v8h*)&af0[ks])[1] = p0[2];
        ((v8h*)&af1[ks])[0] = p1[0]; ((v8h*)&af1[ks])[1] = p1[2];
    }

    float acc0[8], acc1[8];
    #pragma unroll
    for (int r = 0; r < 8; ++r) { acc0[r] = 0.f; acc1[r] = 0.f; }

    for (int nt = 0; nt < NTN; ++nt) {
        const _Float16* bRow = sB + (nt * 16 + mn) * KPAD;
        v8f c0 = {0.f, 0.f, 0.f, 0.f, 0.f, 0.f, 0.f, 0.f};
        v8f c1 = {0.f, 0.f, 0.f, 0.f, 0.f, 0.f, 0.f, 0.f};
        #pragma unroll
        for (int ks = 0; ks < 5; ++ks) {
            const v8h* bp = (const v8h*)(bRow + ks * 32 + kh * 8);
            v16h bv;
            ((v8h*)&bv)[0] = bp[0]; ((v8h*)&bv)[1] = bp[2];
            c0 = __builtin_amdgcn_wmma_f32_16x16x32_f16(false, af0[ks], false, bv,
                                                        (short)0, c0, false, false);
            c1 = __builtin_amdgcn_wmma_f32_16x16x32_f16(false, af1[ks], false, bv,
                                                        (short)0, c1, false, false);
        }
        // C layout: lane holds n = mn for rows m = r + 8*kh; fuse bias+ELU+dot(Wm2)
        const float2 bwv = bw[nt * 16 + mn];
        #pragma unroll
        for (int r = 0; r < 8; ++r) {
            acc0[r] += eluf_fast(c0[r] + bwv.x) * bwv.y;
            acc1[r] += eluf_fast(c1[r] + bwv.x) * bwv.y;
        }
    }

    // reduce over the 16 lanes of each half-wave (each half covers rows 0-7 / 8-15)
    #pragma unroll
    for (int r = 0; r < 8; ++r) {
        for (int off = 8; off > 0; off >>= 1) {
            acc0[r] += __shfl_xor(acc0[r], off, 16);
            acc1[r] += __shfl_xor(acc1[r], off, 16);
        }
    }

    if (mn == 0) {
        const float b2 = bm2[0];
        int base = tile * 32 + kh * 8;
        #pragma unroll
        for (int r = 0; r < 8; ++r) {
            int e0 = base + r;
            int e1 = base + 16 + r;
            if (e0 < nE) out[e0] = acc0[r] + b2;
            if (e1 < nE) out[e1] = acc1[r] + b2;
        }
    }
}

// ---------- host ----------
extern "C" void kernel_launch(void* const* d_in, const int* in_sizes, int n_in,
                              void* d_out, int out_size, void* d_ws, size_t ws_size,
                              hipStream_t stream) {
    const float* pos   = (const float*)d_in[0];
    const int*   eidx  = (const int*)d_in[1];
    const int*   gidx  = (const int*)d_in[2];
    const float* eattr = (const float*)d_in[3];
    const float* W0  = (const float*)d_in[4];
    const float* as0 = (const float*)d_in[5];
    const float* ad0 = (const float*)d_in[6];
    const float* b0  = (const float*)d_in[7];
    const float* W1  = (const float*)d_in[8];
    const float* as1 = (const float*)d_in[9];
    const float* ad1 = (const float*)d_in[10];
    const float* b1  = (const float*)d_in[11];
    const float* Wm1 = (const float*)d_in[12];
    const float* bm1 = (const float*)d_in[13];
    const float* Wm2 = (const float*)d_in[14];
    const float* bm2 = (const float*)d_in[15];
    float* out = (float*)d_out;

    char* ws = (char*)d_ws;
    size_t off = 0;
    auto carve = [&](size_t bytes) -> char* {
        char* p = ws + off;
        off = (off + bytes + 255) & ~(size_t)255;
        return p;
    };
    float*     bufH  = (float*)carve((size_t)NN * 64 * 4);
    float*     bufO  = (float*)carve((size_t)NN * 64 * 4);
    float*     bufX  = (float*)carve((size_t)NN * 64 * 4);
    float*     a_s   = (float*)carve((size_t)NN * 4);
    float*     a_d   = (float*)carve((size_t)NN * 4);
    float*     amax  = (float*)carve((size_t)NN * 4);
    float*     denom = (float*)carve((size_t)NN * 4);
    float*     ew    = (float*)carve((size_t)EGG * 4);
    _Float16*  w1t   = (_Float16*)carve((size_t)NPAD * KPAD * 2);
    float2*    bw    = (float2*)carve((size_t)NPAD * 8);
    (void)in_sizes; (void)n_in; (void)out_size; (void)ws_size;

    const int* gsrc = gidx;          const int* gdst = gidx + EGG;
    const int* esrc = eidx;          const int* edst = eidx + EE;

    prep_weights<<<(NPAD * KPAD + 255) / 256, 256, 0, stream>>>(Wm1, bm1, Wm2, w1t, bw);

    // ---- GAT layer 0 ----
    init_agg<<<(NN * 64 + 255) / 256, 256, 0, stream>>>(bufO, amax, denom);
    gat0_xform<<<(NN + 255) / 256, 256, 0, stream>>>(pos, W0, as0, ad0, bufH, a_s, a_d);
    edge_alpha_max<<<(EGG + 255) / 256, 256, 0, stream>>>(a_s, a_d, gsrc, gdst, amax);
    amax_fix<<<(NN + 255) / 256, 256, 0, stream>>>(amax);
    edge_expsum<<<(EGG + 255) / 256, 256, 0, stream>>>(a_s, a_d, gsrc, gdst, amax, ew, denom);
    edge_agg<<<(EGG * 64 + 255) / 256, 256, 0, stream>>>(gsrc, gdst, ew, denom, bufH, bufO);
    finish_elu<<<(NN * 64 + 255) / 256, 256, 0, stream>>>(bufO, b0, bufX);

    // ---- GAT layer 1 ----
    gat1_xform<<<(NN + 3) / 4, 256, 0, stream>>>(bufX, W1, bufH);
    node_att<<<(NN + 255) / 256, 256, 0, stream>>>(bufH, as1, ad1, a_s, a_d);
    init_agg<<<(NN * 64 + 255) / 256, 256, 0, stream>>>(bufO, amax, denom);
    edge_alpha_max<<<(EGG + 255) / 256, 256, 0, stream>>>(a_s, a_d, gsrc, gdst, amax);
    amax_fix<<<(NN + 255) / 256, 256, 0, stream>>>(amax);
    edge_expsum<<<(EGG + 255) / 256, 256, 0, stream>>>(a_s, a_d, gsrc, gdst, amax, ew, denom);
    edge_agg<<<(EGG * 64 + 255) / 256, 256, 0, stream>>>(gsrc, gdst, ew, denom, bufH, bufO);
    finish_elu<<<(NN * 64 + 255) / 256, 256, 0, stream>>>(bufO, b1, bufX);

    // ---- fused edge MLP (WMMA) ----
    int nTiles = (EE + 31) / 32;             // 18750
    int grid   = (nTiles + 7) / 8;           // 2344
    size_t shmem = (size_t)NPAD * KPAD * 2 + (size_t)8 * 32 * KPAD * 2;  // ~245 KB < 320 KB/WGP
    edge_mlp_wmma<<<grid, 256, shmem, stream>>>(bufX, esrc, edst, eattr, w1t, bw,
                                                bm2, out, EE, nTiles);
}
